// SpatioTemporalAttention_17205638988343
// MI455X (gfx1250) — matrix operand order repaired and checked by reading, not compiled, not run on hardware
//
#include <hip/hip_runtime.h>
#include <math.h>

typedef __bf16 bf16_t;
typedef __attribute__((ext_vector_type(16))) __bf16 bf16x16;
typedef __attribute__((ext_vector_type(8)))  float  f32x8;

#define C_DIM      512
#define N_HEADS    8
#define D_HEAD     64
#define NTOK       16384        // B*F*H*W = 2*8*32*32
#define FF_IN      1365
#define FF_PAD     1376         // FF_IN padded to multiple of 32
#define FF2        2730
#define CPB        256
#define ATTN_SCALE 0.125f

// ---------------------------------------------------------------- helpers
__device__ __forceinline__ float siluf(float x) { return x / (1.f + __expf(-x)); }

__device__ __forceinline__ float block_reduce_sum256(float v, float* red) {
  int lane = threadIdx.x & 31, wave = threadIdx.x >> 5;
  for (int off = 16; off > 0; off >>= 1) v += __shfl_down(v, off, 32);
  if (lane == 0) red[wave] = v;
  __syncthreads();
  if (threadIdx.x == 0) {
    float s = 0.f;
    for (int i = 0; i < 8; ++i) s += red[i];
    red[0] = s;
  }
  __syncthreads();
  return red[0];
}

// ---- CDNA5 async global -> LDS copy (ASYNCcnt tracked, no VGPR data) ----
__device__ __forceinline__ void async_b128(unsigned ldsOff, const void* gaddr) {
  asm volatile("global_load_async_to_lds_b128 %0, %1, off"
               :: "v"(ldsOff), "v"(gaddr) : "memory");
}
__device__ __forceinline__ void async_wait0() {
  asm volatile("s_wait_asynccnt 0x0" ::: "memory");
}

// ---- WMMA fragment loaders from LDS (layouts per CDNA5 ISA 7.12.2) ----
union FragBF { bf16x16 v; uint4 q[2]; };

// A (16x32 bf16): lane lo holds row M=lo; hi selects K-halves {0..7,16..23} vs {8..15,24..31}
__device__ __forceinline__ bf16x16 frag_a_lds(const bf16_t* base, int stride, int lane) {
  const int lo = lane & 15, hi = lane >> 4;
  FragBF f;
  const bf16_t* p = base + lo * stride + hi * 8;
  f.q[0] = *(const uint4*)p;
  f.q[1] = *(const uint4*)(p + 16);
  return f.v;
}
// B (32x16 bf16) from transposed LDS tile Bt[n][k]: lane lo = column n; hi selects K-halves {0..7,16..23} vs {8..15,24..31}
__device__ __forceinline__ bf16x16 frag_b_lds(const bf16_t* base, int stride, int lane) {
  const int lo = lane & 15, hi = lane >> 4;
  FragBF f;
  const bf16_t* p = base + lo * stride + hi * 8;
  f.q[0] = *(const uint4*)p;
  f.q[1] = *(const uint4*)(p + 16);
  return f.v;
}

#define WMMA_BF16(A, B, C) \
  __builtin_amdgcn_wmma_f32_16x16x32_bf16(false, (A), false, (B), (short)0, (C), false, false)

// ---------------------------------------------------------------- fp32 -> bf16 cast with zero padding
__global__ __launch_bounds__(256) void cast_pad_k(const float* __restrict__ src,
                                                  bf16_t* __restrict__ dst,
                                                  int srcR, int srcC, int dstR, int dstC) {
  size_t i = (size_t)blockIdx.x * 256 + threadIdx.x;
  if (i >= (size_t)dstR * dstC) return;
  int r = (int)(i / dstC), c = (int)(i % dstC);
  dst[i] = (r < srcR && c < srcC) ? (bf16_t)src[(size_t)r * srcC + c] : (bf16_t)0.f;
}

// ---------------------------------------------------------------- layout transposes
__global__ __launch_bounds__(256) void transpose_in_k(const float* __restrict__ x,
                                                      float* __restrict__ xs) {
  size_t i = (size_t)blockIdx.x * 256 + threadIdx.x;
  if (i >= (size_t)NTOK * C_DIM) return;
  int c = (int)(i & (C_DIM - 1));
  size_t tok = i >> 9;
  int hw = (int)(tok & 1023);
  int f = (int)((tok >> 10) & 7);
  int b = (int)(tok >> 13);
  xs[i] = x[(((size_t)b * C_DIM + c) * 8 + f) * 1024 + hw];
}

__global__ __launch_bounds__(256) void transpose_out_k(const float* __restrict__ xs,
                                                       float* __restrict__ out) {
  size_t i = (size_t)blockIdx.x * 256 + threadIdx.x;
  if (i >= (size_t)NTOK * C_DIM) return;
  int hw = (int)(i & 1023);
  size_t r = i >> 10;
  int f = (int)(r & 7); r >>= 3;
  int c = (int)(r & (C_DIM - 1));
  int b = (int)(r >> 9);
  out[i] = xs[(((size_t)b * 8 + f) * 1024 + hw) * C_DIM + c];
}

// ---------------------------------------------------------------- CPB bias MLPs
__global__ __launch_bounds__(256) void cpb2d_k(const float* __restrict__ w1, const float* __restrict__ b1,
                                               const float* __restrict__ w2, const float* __restrict__ b2,
                                               const float* __restrict__ w3, const float* __restrict__ b3,
                                               float* __restrict__ tabT) {
  __shared__ float h1[CPB], h2[CPB];
  const int row = blockIdx.x;                 // 0..3968
  const float dy = (float)(row / 63 - 31);
  const float dx = (float)(row % 63 - 31);
  const int t = threadIdx.x;
  float v = dy * w1[t] + dx * w1[CPB + t] + b1[t];
  h1[t] = siluf(v);
  __syncthreads();
  float a = b2[t];
  for (int j = 0; j < CPB; ++j) a += h1[j] * w2[j * CPB + t];
  h2[t] = siluf(a);
  __syncthreads();
  if (t < N_HEADS) {
    float o = b3[t];
    for (int j = 0; j < CPB; ++j) o += h2[j] * w3[j * N_HEADS + t];
    tabT[t * 3969 + row] = o;
  }
}

__global__ __launch_bounds__(256) void cpb1d_k(const float* __restrict__ w1, const float* __restrict__ b1,
                                               const float* __restrict__ w2, const float* __restrict__ b2,
                                               const float* __restrict__ w3, const float* __restrict__ b3,
                                               float* __restrict__ tabT) {
  __shared__ float h1[CPB], h2[CPB];
  const int row = blockIdx.x;                 // 0..14
  const float df = (float)(row - 7);
  const int t = threadIdx.x;
  h1[t] = siluf(df * w1[t] + b1[t]);
  __syncthreads();
  float a = b2[t];
  for (int j = 0; j < CPB; ++j) a += h1[j] * w2[j * CPB + t];
  h2[t] = siluf(a);
  __syncthreads();
  if (t < N_HEADS) {
    float o = b3[t];
    for (int j = 0; j < CPB; ++j) o += h2[j] * w3[j * N_HEADS + t];
    tabT[t * 15 + row] = o;
  }
}

// ---------------------------------------------------------------- RMSNorm (gamma) -> bf16
__global__ __launch_bounds__(256) void rmsnorm_cast_k(const float* __restrict__ x,
                                                      const float* __restrict__ g,
                                                      bf16_t* __restrict__ out) {
  __shared__ float red[8];
  const size_t row = blockIdx.x;
  const float* xr = x + row * C_DIM;
  float ss = 0.f;
  for (int c = threadIdx.x; c < C_DIM; c += 256) { float v = xr[c]; ss += v * v; }
  float tot = block_reduce_sum256(ss, red);
  float inv = 22.627417f / fmaxf(sqrtf(tot), 1e-12f);   // sqrt(512)
  for (int c = threadIdx.x; c < C_DIM; c += 256)
    out[row * C_DIM + c] = (bf16_t)(xr[c] * inv * g[c]);
}

// ---------------------------------------------------------------- generic bf16 WMMA GEMM
// C[M,N] = A[M,K](lda=K) * B[K,N](ldb).  EPI 0: store bf16.  EPI 1: out fp32 = acc + resid.
// Double-buffered LDS; A tile staged with async global->LDS, B tile staged transposed.
template <int EPI>
__global__ __launch_bounds__(256) void gemm_bf16_k(const bf16_t* __restrict__ A,
                                                   const bf16_t* __restrict__ B,
                                                   bf16_t* __restrict__ outB,
                                                   float* __restrict__ outF,
                                                   const float* __restrict__ resid,
                                                   int M, int N, int K, int ldb, int ldo) {
  __shared__ bf16_t As[2][128 * 40];
  __shared__ bf16_t Bt[2][128 * 40];
  const int n0 = blockIdx.x * 128, m0 = blockIdx.y * 128;
  const int t = threadIdx.x, lane = t & 31, wave = t >> 5;
  const int mW = (wave & 3) * 32, nW = (wave >> 2) * 64;

  f32x8 acc[8];
  for (int i = 0; i < 8; ++i)
    for (int j = 0; j < 8; ++j) acc[i][j] = 0.f;

  const int ar = t >> 1, ac = (t & 1) * 16;
  const int br = t >> 3, bc = (t & 7) * 16;

  auto stage = [&](int buf, int k0) {
    { // A tile 128x32: async copy (row-major, 32B per thread = 2 x b128)
      const bf16_t* ga = A + (size_t)(m0 + ar) * K + k0 + ac;
      unsigned lofs = (unsigned)(size_t)&As[buf][ar * 40 + ac];
      async_b128(lofs, ga);
      async_b128(lofs + 16, ga + 8);
    }
    { // B tile 32x128, transposed into LDS: Bt[n][k]
      const int gn = n0 + bc;
      const bf16_t* src = B + (size_t)(k0 + br) * ldb + gn;
      union { bf16_t e[16]; uint4 q[2]; } tmp;
      if (gn + 16 <= N) {
        tmp.q[0] = ((const uint4*)src)[0];
        tmp.q[1] = ((const uint4*)src)[1];
      } else {
        for (int j = 0; j < 16; ++j) tmp.e[j] = (gn + j < N) ? src[j] : (bf16_t)0.f;
      }
      for (int j = 0; j < 16; ++j) Bt[buf][(bc + j) * 40 + br] = tmp.e[j];
      if (k0 + 32 < K)
        __builtin_prefetch(B + (size_t)(k0 + 32 + br) * ldb + gn, 0, 1);
    }
  };

  stage(0, 0);
  async_wait0();
  __syncthreads();

  int cur = 0;
  for (int k0 = 0; k0 < K; k0 += 32) {
    if (k0 + 32 < K) stage(cur ^ 1, k0 + 32);   // overlap next-tile staging with WMMA

    bf16x16 a0 = frag_a_lds(&As[cur][mW * 40], 40, lane);
    bf16x16 a1 = frag_a_lds(&As[cur][(mW + 16) * 40], 40, lane);
    for (int nf = 0; nf < 4; ++nf) {
      bf16x16 bfr = frag_b_lds(&Bt[cur][(nW + 16 * nf) * 40], 40, lane);
      acc[nf]     = WMMA_BF16(a0, bfr, acc[nf]);
      acc[4 + nf] = WMMA_BF16(a1, bfr, acc[4 + nf]);
    }
    async_wait0();
    __syncthreads();
    cur ^= 1;
  }

  const int hi = lane >> 4, lo = lane & 15;
  for (int mf = 0; mf < 2; ++mf)
    for (int nf = 0; nf < 4; ++nf)
      for (int r = 0; r < 8; ++r) {
        int row = m0 + mW + 16 * mf + r + 8 * hi;
        int col = n0 + nW + 16 * nf + lo;
        if (col < N && row < M) {
          float v = acc[mf * 4 + nf][r];
          if (EPI == 0) outB[(size_t)row * ldo + col] = (bf16_t)v;
          else          outF[(size_t)row * ldo + col] = v + resid[(size_t)row * ldo + col];
        }
      }
}

// ---------------------------------------------------------------- GEGLU GEMM: Y = (A@Wa) * gelu(A@Wg)
__global__ __launch_bounds__(256) void gemm_geglu_k(const bf16_t* __restrict__ A,
                                                    const bf16_t* __restrict__ W,
                                                    bf16_t* __restrict__ Y, int M, int K) {
  __shared__ bf16_t As[2][128 * 40];
  __shared__ bf16_t Ba[2][128 * 40];
  __shared__ bf16_t Bg[2][128 * 40];
  const int n0 = blockIdx.x * 128, m0 = blockIdx.y * 128;
  const int t = threadIdx.x, lane = t & 31, wave = t >> 5;
  const int mW = (wave & 3) * 32, nW = (wave >> 2) * 64;

  f32x8 aa[8], ag[8];
  for (int i = 0; i < 8; ++i)
    for (int j = 0; j < 8; ++j) { aa[i][j] = 0.f; ag[i][j] = 0.f; }

  const int ar = t >> 1, ac = (t & 1) * 16;
  const int br = t >> 3, bc = (t & 7) * 16;

  auto stage = [&](int buf, int k0) {
    {
      const bf16_t* ga = A + (size_t)(m0 + ar) * K + k0 + ac;
      unsigned lofs = (unsigned)(size_t)&As[buf][ar * 40 + ac];
      async_b128(lofs, ga);
      async_b128(lofs + 16, ga + 8);
    }
    const int gn = n0 + bc;
    for (int g = 0; g < 2; ++g) {
      const bf16_t* src = W + (size_t)(k0 + br) * FF2 + g * FF_IN + gn;
      bf16_t* dst = g ? Bg[buf] : Ba[buf];
      union { bf16_t e[16]; uint4 q[2]; } tmp;
      if (gn + 16 <= FF_IN) {
        tmp.q[0] = ((const uint4*)src)[0];
        tmp.q[1] = ((const uint4*)src)[1];
      } else {
        for (int j = 0; j < 16; ++j) tmp.e[j] = (gn + j < FF_IN) ? src[j] : (bf16_t)0.f;
      }
      for (int j = 0; j < 16; ++j) dst[(bc + j) * 40 + br] = tmp.e[j];
    }
  };

  stage(0, 0);
  async_wait0();
  __syncthreads();

  int cur = 0;
  for (int k0 = 0; k0 < K; k0 += 32) {
    if (k0 + 32 < K) stage(cur ^ 1, k0 + 32);

    bf16x16 a0 = frag_a_lds(&As[cur][mW * 40], 40, lane);
    bf16x16 a1 = frag_a_lds(&As[cur][(mW + 16) * 40], 40, lane);
    for (int nf = 0; nf < 4; ++nf) {
      bf16x16 ba = frag_b_lds(&Ba[cur][(nW + 16 * nf) * 40], 40, lane);
      bf16x16 bg = frag_b_lds(&Bg[cur][(nW + 16 * nf) * 40], 40, lane);
      aa[nf]     = WMMA_BF16(a0, ba, aa[nf]);
      aa[4 + nf] = WMMA_BF16(a1, ba, aa[4 + nf]);
      ag[nf]     = WMMA_BF16(a0, bg, ag[nf]);
      ag[4 + nf] = WMMA_BF16(a1, bg, ag[4 + nf]);
    }
    async_wait0();
    __syncthreads();
    cur ^= 1;
  }

  const int hi = lane >> 4, lo = lane & 15;
  for (int mf = 0; mf < 2; ++mf)
    for (int nf = 0; nf < 4; ++nf)
      for (int r = 0; r < 8; ++r) {
        int row = m0 + mW + 16 * mf + r + 8 * hi;
        int col = n0 + nW + 16 * nf + lo;
        if (col < FF_IN && row < M) {
          float a = aa[mf * 4 + nf][r];
          float g = ag[mf * 4 + nf][r];
          float gel = 0.5f * g * (1.f + erff(g * 0.70710678f));
          Y[(size_t)row * FF_PAD + col] = (bf16_t)(a * gel);
        }
      }
}

// ---------------------------------------------------------------- spatial flash attention (WMMA)
// grid: x = q-tile (16), y = head (8), z = batch*frame (16)
__global__ __launch_bounds__(256) void spatial_attn_k(const bf16_t* __restrict__ Q,
                                                      const bf16_t* __restrict__ KV,
                                                      const float* __restrict__ tab2,
                                                      bf16_t* __restrict__ O) {
  __shared__ bf16_t Qs[64 * 72];
  __shared__ bf16_t Kt[64 * 72];
  __shared__ bf16_t Vt[64 * 72];
  __shared__ bf16_t Ps[64 * 72];
  __shared__ float  Ss[64 * 68];
  __shared__ float  m_run[64], l_run[64], alpha_s[64];

  const int qt = blockIdx.x, h = blockIdx.y, bfi = blockIdx.z;
  const int t = threadIdx.x, lane = t & 31, wave = t >> 5;
  const int amW = (wave & 3) * 16, anW = (wave >> 2) * 32;
  const int hi = lane >> 4, lo = lane & 15;
  const size_t rowq0 = (size_t)bfi * 1024 + qt * 64;
  const int r = t >> 2, c = (t & 3) * 16;

  { // load Q tile (64 x 64) for this head -- async copy
    const bf16_t* qsrc = Q + (rowq0 + r) * C_DIM + h * D_HEAD + c;
    unsigned qofs = (unsigned)(size_t)&Qs[r * 72 + c];
    async_b128(qofs, qsrc);
    async_b128(qofs + 16, qsrc + 8);
  }
  if (t < 64) { m_run[t] = -1e30f; l_run[t] = 0.f; }

  f32x8 oacc[2];
  for (int f = 0; f < 2; ++f)
    for (int rr = 0; rr < 8; ++rr) oacc[f][rr] = 0.f;

  for (int jt = 0; jt < 16; ++jt) {
    const int jbase = jt * 64;
    { // K tile: async copy (row-major == B-transposed layout); V tile: manual transpose
      const size_t krow = ((size_t)bfi * 1024 + jbase + r) * 1024;
      const bf16_t* ksrc = KV + krow + h * D_HEAD + c;
      unsigned kofs = (unsigned)(size_t)&Kt[r * 72 + c];
      async_b128(kofs, ksrc);
      async_b128(kofs + 16, ksrc + 8);
      union { bf16_t e[16]; uint4 q[2]; } tmp;
      const uint4* vs = (const uint4*)(KV + krow + C_DIM + h * D_HEAD + c);
      tmp.q[0] = vs[0];
      tmp.q[1] = vs[1];
      for (int j = 0; j < 16; ++j) Vt[(c + j) * 72 + r] = tmp.e[j];
    }
    async_wait0();
    __syncthreads();

    // S = Q * K^T
    f32x8 sacc[2];
    for (int f = 0; f < 2; ++f)
      for (int rr = 0; rr < 8; ++rr) sacc[f][rr] = 0.f;
    for (int ks = 0; ks < 64; ks += 32) {
      bf16x16 afr = frag_a_lds(Qs + amW * 72 + ks, 72, lane);
      for (int f = 0; f < 2; ++f) {
        bf16x16 bfr = frag_b_lds(Kt + (anW + 16 * f) * 72 + ks, 72, lane);
        sacc[f] = WMMA_BF16(afr, bfr, sacc[f]);
      }
    }
    for (int f = 0; f < 2; ++f)
      for (int rr = 0; rr < 8; ++rr)
        Ss[(amW + rr + 8 * hi) * 68 + anW + 16 * f + lo] = sacc[f][rr];
    __syncthreads();

    // online softmax, one thread per row
    if (t < 64) {
      const int row = t;
      const int ig = qt * 64 + row;
      const int yi = ig >> 5, xi = ig & 31;
      float rowmax = m_run[row];
      for (int cc = 0; cc < 64; ++cc) {
        const int jg = jbase + cc;
        const int idx = (yi - (jg >> 5) + 31) * 63 + (xi - (jg & 31) + 31);
        float s = Ss[row * 68 + cc] * ATTN_SCALE + tab2[h * 3969 + idx];
        Ss[row * 68 + cc] = s;
        rowmax = fmaxf(rowmax, s);
      }
      const float alpha = __expf(m_run[row] - rowmax);
      float lsum = 0.f;
      for (int cc = 0; cc < 64; ++cc) {
        float pp = __expf(Ss[row * 68 + cc] - rowmax);
        Ps[row * 72 + cc] = (bf16_t)pp;
        lsum += pp;
      }
      l_run[row] = l_run[row] * alpha + lsum;
      m_run[row] = rowmax;
      alpha_s[row] = alpha;
    }
    __syncthreads();

    // rescale running O, then O += P * V
    for (int f = 0; f < 2; ++f)
      for (int rr = 0; rr < 8; ++rr) oacc[f][rr] *= alpha_s[amW + rr + 8 * hi];
    for (int ks = 0; ks < 64; ks += 32) {
      bf16x16 afr = frag_a_lds(Ps + amW * 72 + ks, 72, lane);
      for (int f = 0; f < 2; ++f) {
        bf16x16 bfr = frag_b_lds(Vt + (anW + 16 * f) * 72 + ks, 72, lane);
        oacc[f] = WMMA_BF16(afr, bfr, oacc[f]);
      }
    }
    __syncthreads();
  }

  for (int f = 0; f < 2; ++f)
    for (int rr = 0; rr < 8; ++rr) {
      const int row = amW + rr + 8 * hi;
      const int col = anW + 16 * f + lo;
      float ov = oacc[f][rr] / l_run[row];
      O[(rowq0 + row) * C_DIM + h * D_HEAD + col] = (bf16_t)ov;
    }
}

// ---------------------------------------------------------------- temporal attention (N=8, VALU)
__global__ __launch_bounds__(256) void temporal_attn_k(const bf16_t* __restrict__ Q,
                                                       const bf16_t* __restrict__ KV,
                                                       const float* __restrict__ tab1,
                                                       bf16_t* __restrict__ O) {
  __shared__ float Sa[N_HEADS][8][8];
  const int s = blockIdx.x;                    // 0..2047: (b, hw)
  const int b = s >> 10, hw = s & 1023;
  const int t = threadIdx.x, h = t >> 5, lane = t & 31;
  const size_t base = (size_t)b * 8 * 1024 + hw;   // token(f) = base + f*1024
  const size_t cq = (size_t)h * D_HEAD;

  for (int p = lane; p < 64; p += 32) {
    const int fi = p >> 3, fj = p & 7;
    const bf16_t* qp = Q + (base + (size_t)fi * 1024) * C_DIM + cq;
    const bf16_t* kp = KV + (base + (size_t)fj * 1024) * 1024 + cq;
    float acc = 0.f;
    for (int d = 0; d < D_HEAD; ++d) acc += (float)qp[d] * (float)kp[d];
    Sa[h][fi][fj] = acc * ATTN_SCALE + tab1[h * 15 + (fi - fj + 7)];
  }
  __syncthreads();
  if (lane < 8) {
    const int fi = lane;
    float mx = -1e30f;
    for (int fj = 0; fj < 8; ++fj) mx = fmaxf(mx, Sa[h][fi][fj]);
    float sum = 0.f;
    for (int fj = 0; fj < 8; ++fj) { float e = __expf(Sa[h][fi][fj] - mx); Sa[h][fi][fj] = e; sum += e; }
    const float inv = 1.f / sum;
    for (int fj = 0; fj < 8; ++fj) Sa[h][fi][fj] *= inv;
  }
  __syncthreads();
  for (int fi = 0; fi < 8; ++fi)
    for (int dd = 0; dd < 2; ++dd) {
      const int d = lane + dd * 32;
      float acc = 0.f;
      for (int fj = 0; fj < 8; ++fj)
        acc += Sa[h][fi][fj] * (float)KV[(base + (size_t)fj * 1024) * 1024 + C_DIM + cq + d];
      O[(base + (size_t)fi * 1024) * C_DIM + cq + d] = (bf16_t)acc;
    }
}

// ---------------------------------------------------------------- FF shift + RMSNorm -> bf16
__global__ __launch_bounds__(256) void ff_shift_rmsnorm_k(const bf16_t* __restrict__ y,
                                                          const float* __restrict__ g,
                                                          bf16_t* __restrict__ yn) {
  __shared__ float buf[FF_PAD];
  __shared__ float red[8];
  const size_t tok = blockIdx.x;
  const int f = (int)((tok >> 10) & 7);
  float ss = 0.f;
  for (int c = threadIdx.x; c < FF_PAD; c += 256) {
    float v = 0.f;
    if (c < 683)                    v = (float)y[tok * FF_PAD + c];
    else if (c < FF_IN && f > 0)    v = (float)y[(tok - 1024) * FF_PAD + c];
    buf[c] = v;
    ss += v * v;
  }
  float tot = block_reduce_sum256(ss, red);
  float inv = 36.945906f / fmaxf(sqrtf(tot), 1e-12f);   // sqrt(1365)
  for (int c = threadIdx.x; c < FF_PAD; c += 256)
    yn[tok * FF_PAD + c] = (bf16_t)((c < FF_IN) ? buf[c] * inv * g[c] : 0.f);
}

// ================================================================ host launch
extern "C" void kernel_launch(void* const* d_in, const int* in_sizes, int n_in,
                              void* d_out, int out_size, void* d_ws, size_t ws_size,
                              hipStream_t stream) {
  (void)in_sizes; (void)n_in; (void)out_size; (void)ws_size;
  const float* x        = (const float*)d_in[0];
  const float* sa_gamma = (const float*)d_in[1];
  const float* sa_wq    = (const float*)d_in[2];
  const float* sa_wkv   = (const float*)d_in[3];
  const float* sa_wo    = (const float*)d_in[4];
  const float* ta_gamma = (const float*)d_in[5];
  const float* ta_wq    = (const float*)d_in[6];
  const float* ta_wkv   = (const float*)d_in[7];
  const float* ta_wo    = (const float*)d_in[8];
  const float* sp_w1 = (const float*)d_in[9];
  const float* sp_b1 = (const float*)d_in[10];
  const float* sp_w2 = (const float*)d_in[11];
  const float* sp_b2 = (const float*)d_in[12];
  const float* sp_w3 = (const float*)d_in[13];
  const float* sp_b3 = (const float*)d_in[14];
  const float* tp_w1 = (const float*)d_in[15];
  const float* tp_b1 = (const float*)d_in[16];
  const float* tp_w2 = (const float*)d_in[17];
  const float* tp_b2 = (const float*)d_in[18];
  const float* tp_w3 = (const float*)d_in[19];
  const float* tp_b3 = (const float*)d_in[20];
  const float* ff_win   = (const float*)d_in[21];
  const float* ff_gamma = (const float*)d_in[22];
  const float* ff_wout  = (const float*)d_in[23];

  char* p = (char*)d_ws;
  auto alloc = [&](size_t bytes) -> char* {
    char* r = p;
    p += (bytes + 255) & ~(size_t)255;
    return r;
  };
  float*  xs  = (float*)alloc((size_t)NTOK * C_DIM * 4);
  bf16_t* xn  = (bf16_t*)alloc((size_t)NTOK * C_DIM * 2);
  bf16_t* qb  = (bf16_t*)alloc((size_t)NTOK * C_DIM * 2);
  bf16_t* kvb = (bf16_t*)alloc((size_t)NTOK * 1024 * 2);
  bf16_t* ob  = (bf16_t*)alloc((size_t)NTOK * C_DIM * 2);
  bf16_t* wq_s  = (bf16_t*)alloc((size_t)512 * 512 * 2);
  bf16_t* wkv_s = (bf16_t*)alloc((size_t)512 * 1024 * 2);
  bf16_t* wo_s  = (bf16_t*)alloc((size_t)512 * 512 * 2);
  bf16_t* wq_t  = (bf16_t*)alloc((size_t)512 * 512 * 2);
  bf16_t* wkv_t = (bf16_t*)alloc((size_t)512 * 1024 * 2);
  bf16_t* wo_t  = (bf16_t*)alloc((size_t)512 * 512 * 2);
  bf16_t* win_b  = (bf16_t*)alloc((size_t)512 * FF2 * 2);
  bf16_t* wout_b = (bf16_t*)alloc((size_t)FF_PAD * 512 * 2);
  float* tab2 = (float*)alloc((size_t)8 * 3969 * 4);
  float* tab1 = (float*)alloc((size_t)8 * 15 * 4);
  bf16_t* yb  = (bf16_t*)alloc((size_t)NTOK * FF_PAD * 2);
  bf16_t* ynb = (bf16_t*)alloc((size_t)NTOK * FF_PAD * 2);

  auto castw = [&](const float* src, bf16_t* dst, int sr, int sc, int dr, int dc) {
    size_t n = (size_t)dr * dc;
    cast_pad_k<<<dim3((unsigned)((n + 255) / 256)), dim3(256), 0, stream>>>(src, dst, sr, sc, dr, dc);
  };

  // weight conversion (fp32 -> bf16, with K padding for ff_wout)
  castw(sa_wq, wq_s, 512, 512, 512, 512);
  castw(sa_wkv, wkv_s, 512, 1024, 512, 1024);
  castw(sa_wo, wo_s, 512, 512, 512, 512);
  castw(ta_wq, wq_t, 512, 512, 512, 512);
  castw(ta_wkv, wkv_t, 512, 1024, 512, 1024);
  castw(ta_wo, wo_t, 512, 512, 512, 512);
  castw(ff_win, win_b, 512, FF2, 512, FF2);
  castw(ff_wout, wout_b, FF_IN, 512, FF_PAD, 512);

  // relative position bias tables
  cpb2d_k<<<dim3(3969), dim3(256), 0, stream>>>(sp_w1, sp_b1, sp_w2, sp_b2, sp_w3, sp_b3, tab2);
  cpb1d_k<<<dim3(15), dim3(256), 0, stream>>>(tp_w1, tp_b1, tp_w2, tp_b2, tp_w3, tp_b3, tab1);

  // (b,c,f,h,w) -> token-major (b,f,h,w,c)
  const unsigned ELEM_BLKS = (unsigned)(((size_t)NTOK * C_DIM + 255) / 256);
  transpose_in_k<<<dim3(ELEM_BLKS), dim3(256), 0, stream>>>(x, xs);

  // ---------------- spatial attention ----------------
  rmsnorm_cast_k<<<dim3(NTOK), dim3(256), 0, stream>>>(xs, sa_gamma, xn);
  gemm_bf16_k<0><<<dim3(4, 128), dim3(256), 0, stream>>>(xn, wq_s, qb, nullptr, nullptr,
                                                         NTOK, 512, 512, 512, 512);
  gemm_bf16_k<0><<<dim3(8, 128), dim3(256), 0, stream>>>(xn, wkv_s, kvb, nullptr, nullptr,
                                                         NTOK, 1024, 512, 1024, 1024);
  spatial_attn_k<<<dim3(16, 8, 16), dim3(256), 0, stream>>>(qb, kvb, tab2, ob);
  gemm_bf16_k<1><<<dim3(4, 128), dim3(256), 0, stream>>>(ob, wo_s, nullptr, xs, xs,
                                                         NTOK, 512, 512, 512, 512);

  // ---------------- temporal attention ----------------
  rmsnorm_cast_k<<<dim3(NTOK), dim3(256), 0, stream>>>(xs, ta_gamma, xn);
  gemm_bf16_k<0><<<dim3(4, 128), dim3(256), 0, stream>>>(xn, wq_t, qb, nullptr, nullptr,
                                                         NTOK, 512, 512, 512, 512);
  gemm_bf16_k<0><<<dim3(8, 128), dim3(256), 0, stream>>>(xn, wkv_t, kvb, nullptr, nullptr,
                                                         NTOK, 1024, 512, 1024, 1024);
  temporal_attn_k<<<dim3(2048), dim3(256), 0, stream>>>(qb, kvb, tab1, ob);
  gemm_bf16_k<1><<<dim3(4, 128), dim3(256), 0, stream>>>(ob, wo_t, nullptr, xs, xs,
                                                         NTOK, 512, 512, 512, 512);

  // ---------------- gated feed-forward ----------------
  castw(xs, xn, NTOK, 512, NTOK, 512);   // plain cast (no pre-norm in reference FF)
  hipMemsetAsync(yb, 0, (size_t)NTOK * FF_PAD * 2, stream);
  gemm_geglu_k<<<dim3(11, 128), dim3(256), 0, stream>>>(xn, win_b, yb, NTOK, 512);
  ff_shift_rmsnorm_k<<<dim3(NTOK), dim3(256), 0, stream>>>(yb, ff_gamma, ynb);
  gemm_bf16_k<1><<<dim3(4, 128), dim3(256), 0, stream>>>(ynb, wout_b, nullptr, xs, xs,
                                                         NTOK, 512, FF_PAD, 512, 512);

  // token-major -> (b,c,f,h,w)
  transpose_out_k<<<dim3(ELEM_BLKS), dim3(256), 0, stream>>>(xs, (float*)d_out);
}
